// SelfAttention_39960375722336
// MI455X (gfx1250) — matrix operand order repaired and checked by reading, not compiled, not run on hardware
//
#include <hip/hip_runtime.h>

typedef _Float16 v8h  __attribute__((ext_vector_type(8)));
typedef _Float16 v16h __attribute__((ext_vector_type(16)));
typedef float    v8f  __attribute__((ext_vector_type(8)));
typedef unsigned int u32x4 __attribute__((ext_vector_type(4)));
typedef int          i32x4 __attribute__((ext_vector_type(4)));
typedef int          i32x8 __attribute__((ext_vector_type(8)));

constexpr int B_ = 8;
constexpr int C_ = 256;
constexpr int N_ = 4096;   // H*W
constexpr int D_ = 32;     // CQK

// ---------------------------------------------------------------------------
// Single-instruction max: avoid clang's canonicalizing fmaxf expansion
// (max(max(a,a), max(b,b))).  v_max_num_f32 is IEEE maximumNumber already.
// ---------------------------------------------------------------------------
__device__ __forceinline__ float vmaxf(float a, float b) {
  float r;
  asm("v_max_num_f32 %0, %1, %2" : "=v"(r) : "v"(a), "v"(b));
  return r;
}

// ---------------------------------------------------------------------------
// Fused DPP16 butterfly reductions across the 16 lanes of each half-wave
// (wave32).  DPP is applied as a src0 modifier of the max/add itself, so each
// butterfly stage is exactly ONE VALU instruction.
// Stages: quad_perm xor1, quad_perm xor2, row_half_mirror, row_mirror.
// ---------------------------------------------------------------------------
__device__ __forceinline__ float red_max16(float v) {
  float t;
  asm("v_max_num_f32_dpp %0, %1, %1 quad_perm:[1,0,3,2] row_mask:0xf bank_mask:0xf bound_ctrl:1"
      : "=v"(t) : "v"(v));
  asm("v_max_num_f32_dpp %0, %1, %1 quad_perm:[2,3,0,1] row_mask:0xf bank_mask:0xf bound_ctrl:1"
      : "=v"(v) : "v"(t));
  asm("v_max_num_f32_dpp %0, %1, %1 row_half_mirror row_mask:0xf bank_mask:0xf bound_ctrl:1"
      : "=v"(t) : "v"(v));
  asm("v_max_num_f32_dpp %0, %1, %1 row_mirror row_mask:0xf bank_mask:0xf bound_ctrl:1"
      : "=v"(v) : "v"(t));
  return v;
}
__device__ __forceinline__ float red_sum16(float v) {
  float t;
  asm("v_add_f32_dpp %0, %1, %1 quad_perm:[1,0,3,2] row_mask:0xf bank_mask:0xf bound_ctrl:1"
      : "=v"(t) : "v"(v));
  asm("v_add_f32_dpp %0, %1, %1 quad_perm:[2,3,0,1] row_mask:0xf bank_mask:0xf bound_ctrl:1"
      : "=v"(v) : "v"(t));
  asm("v_add_f32_dpp %0, %1, %1 row_half_mirror row_mask:0xf bank_mask:0xf bound_ctrl:1"
      : "=v"(t) : "v"(v));
  asm("v_add_f32_dpp %0, %1, %1 row_mirror row_mask:0xf bank_mask:0xf bound_ctrl:1"
      : "=v"(v) : "v"(t));
  return v;
}

// ---------------------------------------------------------------------------
// Tensor Data Mover: load a [256 rows x 32 cols] f16 tile of Vt (row stride
// N_) into LDS with rows packed at 64B -> shV[c*32 + j] layout.
// D# group0: count=1, lds_addr, global_addr, type=2.
// D# group1: data_size=2B, tensor_dim0=N_, tensor_dim1=C_, tile_dim0=32,
//            tile_dim1=C_, tensor_dim0_stride=N_.
// Groups 2/3 unused (2-D tensor) -> zeros.  6-arg clang-23 builtin form.
// ---------------------------------------------------------------------------
__device__ __forceinline__ void tdm_load_v_tile(const _Float16* gsrc,
                                                unsigned lds_off) {
  const unsigned long long ga = (unsigned long long)gsrc;
  u32x4 g0;
  g0[0] = 0x1u;                                        // count=1, user mode
  g0[1] = lds_off;                                     // lds_addr (bytes)
  g0[2] = (unsigned)ga;                                // global_addr[31:0]
  g0[3] = (unsigned)((ga >> 32) & 0x1FFFFFFu)          // global_addr[56:32]
        | (2u << 30);                                  // type=2 (image)
  i32x8 g1;
  g1[0] = (int)(1u << 16);                             // data_size=1 (2 bytes)
  g1[1] = (int)(((unsigned)N_ & 0xFFFFu) << 16);       // tensor_dim0[15:0]
  g1[2] = (int)(((unsigned)N_ >> 16)
        | (((unsigned)C_ & 0xFFFFu) << 16));           // dim0[31:16]|dim1[15:0]
  g1[3] = (int)(((unsigned)C_ >> 16) | (32u << 16));   // dim1[31:16]|tile_dim0
  g1[4] = (int)(unsigned)C_;                           // tile_dim1=256, tile_dim2=0
  g1[5] = (int)(unsigned)N_;                           // tensor_dim0_stride lo
  g1[6] = 0;                                           // stride hi | dim1_stride lo
  g1[7] = 0;
  const i32x4 z4 = {0, 0, 0, 0};
  const i32x8 z8 = {0, 0, 0, 0, 0, 0, 0, 0};
  __builtin_amdgcn_tensor_load_to_lds(g0, g1, z4, z4, z8, 0);
}

// ---------------------------------------------------------------------------
// Kernel 1: fused QKV 1x1-conv projection.
// blockIdx.y selects a 32-output chunk: 0 -> Q, 1 -> K, 2..9 -> V rows.
// Q,K stored [b, n, 32] f16; V stored transposed [b, c, n] f16 so that
// stage-2 B-operand/TDM reads are contiguous and stage-1 stores coalesce.
// ---------------------------------------------------------------------------
__global__ __launch_bounds__(256) void qkv_proj_kernel(
    const float* __restrict__ x,
    const float* __restrict__ wq, const float* __restrict__ bq,
    const float* __restrict__ wk, const float* __restrict__ bk,
    const float* __restrict__ wv, const float* __restrict__ bv,
    _Float16* __restrict__ Qh, _Float16* __restrict__ Kh,
    _Float16* __restrict__ Vt)
{
  __shared__ float sw[32 * C_];   // 32 KB of weights for this chunk
  const int chunk = blockIdx.y;
  const float* w; const float* bias; int o0 = 0;
  if (chunk == 0)      { w = wq; bias = bq; }
  else if (chunk == 1) { w = wk; bias = bk; }
  else { o0 = (chunk - 2) * 32; w = wv + (size_t)o0 * C_; bias = bv + o0; }

  for (int i = threadIdx.x; i < 32 * C_; i += blockDim.x) sw[i] = w[i];
  __syncthreads();

  const int gid = blockIdx.x * blockDim.x + threadIdx.x;  // pixel in [0, B*N)
  const int b = gid >> 12;          // / N_
  const int n = gid & (N_ - 1);     // % N_

  float acc[32];
  #pragma unroll
  for (int o = 0; o < 32; ++o) acc[o] = bias[o];

  const float* xp = x + (size_t)b * C_ * N_ + n;
  for (int c = 0; c < C_; ++c) {
    const float xv = xp[(size_t)c * N_];   // coalesced across threads
    #pragma unroll
    for (int o = 0; o < 32; ++o) acc[o] = fmaf(sw[o * C_ + c], xv, acc[o]);
  }

  if (chunk == 0) {
    _Float16* qp = Qh + (size_t)gid * D_;
    #pragma unroll
    for (int o = 0; o < 32; ++o) qp[o] = (_Float16)acc[o];
  } else if (chunk == 1) {
    _Float16* kp = Kh + (size_t)gid * D_;
    #pragma unroll
    for (int o = 0; o < 32; ++o) kp[o] = (_Float16)acc[o];
  } else {
    _Float16* vp = Vt + ((size_t)(b * C_ + o0)) * N_ + n;
    #pragma unroll
    for (int o = 0; o < 32; ++o) vp[(size_t)o * N_] = (_Float16)acc[o];  // coalesced
  }
}

// ---------------------------------------------------------------------------
// Kernel 2: fused flash attention + residual epilogue.
// Block = 256 threads = 8 waves; wave w owns Q rows [i0, i0+16).
// Q is pre-scaled by log2(e) so S comes out of the QK^T WMMA in the exp2
// domain: all softmax exponentials are raw v_exp_f32 (exp2), no per-value
// multiplies.  Per 32-column j-step:
//   S(16x32)  = (Q*log2e) * K^T -> 2x v_wmma_f32_16x16x32_f16
//   online softmax              -> fused DPP16 butterflies + v_exp_f32
//   O(16x256) += P * V          -> 16x v_wmma_f32_16x16x32_f16
// V tiles are DMA'd by the Tensor Data Mover into a double-buffered LDS
// tile (DMA of tile j+1 overlaps compute of tile j); K operands are
// software-pipelined in registers so the QK^T WMMAs never wait on VMEM.
// ---------------------------------------------------------------------------
__global__ __launch_bounds__(256) void flash_attn_kernel(
    const _Float16* __restrict__ Qh, const _Float16* __restrict__ Kh,
    const _Float16* __restrict__ Vt, const float* __restrict__ x,
    const float* __restrict__ gamma, float* __restrict__ out)
{
  __shared__ __attribute__((aligned(64))) _Float16 shV[2][C_ * 32];   // 2 x 16 KB
  __shared__ __attribute__((aligned(64))) _Float16 shP[8][16][32];    // 8 KB

  const int wave = threadIdx.x >> 5;
  const int lane = threadIdx.x & 31;
  const int col  = lane & 15;
  const int hi   = lane >> 4;

  const int b  = blockIdx.y;
  const int i0 = blockIdx.x * 128 + wave * 16;

  const _Float16* Vb = Vt + (size_t)b * C_ * N_;
  const unsigned shv_base   = (unsigned)(unsigned long long)(&shV[0][0]);
  const unsigned shv_stride = (unsigned)(C_ * 32 * sizeof(_Float16));  // 16 KB

  // ---- Q tile as WMMA A operand (16x32 f16), pre-scaled by log2(e) ----
  // A layout: lanes 0-15 row M=lane hold K=0..15; lanes 16-31 row
  // M=lane-16 hold K=16..31.
  const int kbase = hi ? 16 : 0;
  const _Float16* qrow = Qh + ((size_t)b * N_ + i0 + col) * D_;
  v16h qa;
  ((v8h*)&qa)[0] = *(const v8h*)(qrow + kbase);
  ((v8h*)&qa)[1] = *(const v8h*)(qrow + kbase + 8);
  qa = qa * (_Float16)1.44269504f;   // 8x v_pk_mul_f16, once per kernel

  float mrun[8], lrun[8];
  #pragma unroll
  for (int r = 0; r < 8; ++r) { mrun[r] = -3.0e38f; lrun[r] = 0.0f; }

  v8f acc[16];            // 16x256 f32 output tile, 128 VGPRs
  #pragma unroll
  for (int t = 0; t < 16; ++t)
    #pragma unroll
    for (int e = 0; e < 8; ++e) acc[t][e] = 0.0f;

  // B layout: lanes 0-15 column N=lane hold K=0..15; lanes 16-31 same
  // columns hold K=16..31 (16 contiguous halves each).
  const int dbase = hi ? 16 : 0;
  const int jb    = hi ? 16 : 0;

  // K operand address for column block starting at row j (per-lane column)
  const _Float16* Kbase = Kh + ((size_t)b * N_ + col) * D_ + dbase;

  // prologue: DMA first V tile into buffer 0; preload first K operands
  if (wave == 0) tdm_load_v_tile(Vb, shv_base);
  v16h kb0 = *(const v16h*)(Kbase);
  v16h kb1 = *(const v16h*)(Kbase + (size_t)16 * D_);

  for (int j0 = 0; j0 < N_; j0 += 32) {
    const int cur = (j0 >> 5) & 1;
    __syncthreads();  // all reads of the buffer being overwritten are done

    if (wave == 0) {
      if (j0 + 32 < N_) {
        // prefetch next V tile into the other buffer, then wait for current
        tdm_load_v_tile(Vb + (j0 + 32), shv_base + (cur ^ 1) * shv_stride);
        __builtin_amdgcn_s_wait_tensorcnt(1);
      } else {
        __builtin_amdgcn_s_wait_tensorcnt(0);
      }
    }

    v8f zero;
    #pragma unroll
    for (int e = 0; e < 8; ++e) zero[e] = 0.0f;

    v8f s0 = __builtin_amdgcn_wmma_f32_16x16x32_f16(
        false, qa, false, kb0, (short)0, zero, false, false);
    v8f s1 = __builtin_amdgcn_wmma_f32_16x16x32_f16(
        false, qa, false, kb1, (short)0, zero, false, false);

    // issue next iteration's K loads now; they retire during softmax+PV
    // (wraps to j=0 on the final iteration -- harmless dead load, no branch)
    const int jn = (j0 + 32) & (N_ - 1);
    const v16h nk0 = *(const v16h*)(Kbase + (size_t)jn * D_);
    const v16h nk1 = *(const v16h*)(Kbase + (size_t)(jn + 16) * D_);

    // ---- online softmax over these 32 columns (exp2 domain) ----
    // C/D layout: lane holds column col, rows {r+8*hi}; row reductions are
    // single-instruction DPP16 butterflies across each half-wave.
    float p0[8], p1[8], scale[8];
    #pragma unroll
    for (int r = 0; r < 8; ++r) {
      const float m = red_max16(vmaxf(s0[r], s1[r]));
      const float mnew = vmaxf(mrun[r], m);
      const float sc = exp2f(mrun[r] - mnew);
      p0[r] = exp2f(s0[r] - mnew);
      p1[r] = exp2f(s1[r] - mnew);
      const float rs = red_sum16(p0[r] + p1[r]);
      lrun[r] = lrun[r] * sc + rs;
      mrun[r] = mnew;
      scale[r] = sc;
    }

    #pragma unroll
    for (int t = 0; t < 16; ++t)
      #pragma unroll
      for (int r = 0; r < 8; ++r) acc[t][r] *= scale[r];

    // P tile to LDS in C/D layout...
    #pragma unroll
    for (int r = 0; r < 8; ++r) {
      shP[wave][r + 8 * hi][col]      = (_Float16)p0[r];
      shP[wave][r + 8 * hi][col + 16] = (_Float16)p1[r];
    }
    __syncthreads();  // V tile (TDM done, waited by wave 0) + all P visible

    // ...re-read as A operand (16x32 f16)
    v16h pa;
    ((v8h*)&pa)[0] = *(const v8h*)(&shP[wave][col][kbase]);
    ((v8h*)&pa)[1] = *(const v8h*)(&shP[wave][col][kbase + 8]);

    // ---- O += P * V : 16 WMMAs cover all 256 channels ----
    #pragma unroll
    for (int t = 0; t < 16; ++t) {
      const int c = t * 16 + col;
      const v16h vb = *(const v16h*)(&shV[cur][c * 32 + jb]);
      acc[t] = __builtin_amdgcn_wmma_f32_16x16x32_f16(
          false, pa, false, vb, (short)0, acc[t], false, false);
    }

    kb0 = nk0;
    kb1 = nk1;
  }

  // ---- epilogue: out = gamma * (O / l) + x ----
  const float g = gamma[0];
  const size_t base = (size_t)b * C_ * N_;
  #pragma unroll
  for (int r = 0; r < 8; ++r) {
    const float linv = 1.0f / lrun[r];
    const int i = i0 + r + 8 * hi;
    #pragma unroll
    for (int t = 0; t < 16; ++t) {
      const int c = t * 16 + col;
      const size_t idx = base + (size_t)c * N_ + i;
      out[idx] = g * acc[t][r] * linv + x[idx];
    }
  }
}

// ---------------------------------------------------------------------------
extern "C" void kernel_launch(void* const* d_in, const int* in_sizes, int n_in,
                              void* d_out, int out_size, void* d_ws, size_t ws_size,
                              hipStream_t stream) {
  const float* x     = (const float*)d_in[0];
  const float* wq    = (const float*)d_in[1];
  const float* bq    = (const float*)d_in[2];
  const float* wk    = (const float*)d_in[3];
  const float* bk    = (const float*)d_in[4];
  const float* wv    = (const float*)d_in[5];
  const float* bv    = (const float*)d_in[6];
  const float* gamma = (const float*)d_in[7];
  float* out = (float*)d_out;

  char* ws = (char*)d_ws;
  const size_t qk_bytes = (size_t)B_ * N_ * D_ * 2;   // 2 MB each
  _Float16* Qh = (_Float16*)ws;
  _Float16* Kh = (_Float16*)(ws + qk_bytes);
  _Float16* Vt = (_Float16*)(ws + 2 * qk_bytes);      // 16 MB, [b, c, n]

  // Stage 1: QKV projections (10 chunks of 32 outputs: Q, K, 8x V)
  qkv_proj_kernel<<<dim3((B_ * N_) / 256, 10), 256, 0, stream>>>(
      x, wq, bq, wk, bk, wv, bv, Qh, Kh, Vt);

  // Stage 2: fused flash attention + residual (TDM-fed V tiles)
  flash_attn_kernel<<<dim3(N_ / 128, B_), 256, 0, stream>>>(
      Qh, Kh, Vt, x, gamma, out);
}
